// Decoder_16226386444776
// MI455X (gfx1250) — matrix operand order repaired and checked by reading, not compiled, not run on hardware
//
#include <hip/hip_runtime.h>
#include <hip/hip_bf16.h>
#include <string.h>

// Problem dims (fixed by reference)
#define B 64
#define S 512
#define T 128
#define E 256
#define H 512
#define K2H (2*H)           // 1024
#define N4H (4*H)           // 2048
#define KCAT (E + 2*H + H)  // 1792

typedef __attribute__((ext_vector_type(16))) __bf16 v16bf;
typedef __attribute__((ext_vector_type(8)))  float  v8f;

static __device__ __forceinline__ float fast_rcp(float x) {
    return __builtin_amdgcn_rcpf(x);
}
static __device__ __forceinline__ float fast_sigmoid(float x) {
    return fast_rcp(1.0f + __expf(-x));
}
static __device__ __forceinline__ float fast_tanh(float x) {
    float e = __expf(2.0f * x);
    return 1.0f - 2.0f * fast_rcp(e + 1.0f);
}
// exact bf16->f32 unpack of a packed pair (lo = even index, hi = odd index)
static __device__ __forceinline__ float bf_lo(unsigned u) { return __uint_as_float(u << 16); }
static __device__ __forceinline__ float bf_hi(unsigned u) { return __uint_as_float(u & 0xffff0000u); }

// ---------------------------------------------------------------------------
// CDNA5 async global->LDS copy (ASYNCcnt path), inline asm per ISA 15.18.3
// (portable across both toolchains; builtin arity differs between them).
// LDS operand VGPR holds the byte offset from the workgroup LDS base, which is
// exactly what ptrtoint of an addrspace(3) pointer yields.
// ---------------------------------------------------------------------------
static __device__ __forceinline__ void async_load_f32_to_lds(float* lds_dst,
                                                             const float* gsrc) {
    unsigned loff = (unsigned)(uintptr_t)(__attribute__((address_space(3))) float*)lds_dst;
    asm volatile("global_load_async_to_lds_b32 %0, %1, off"
                 :: "v"(loff), "v"(gsrc) : "memory");
}
static __device__ __forceinline__ void wait_asynccnt0() {
    asm volatile("s_wait_asynccnt 0" ::: "memory");
}

// ---------------------------------------------------------------------------
// Fragment loaders (ISA 7.12.2 layouts); raw bf16, contiguous vector loads.
// ---------------------------------------------------------------------------
static __device__ __forceinline__ v16bf load_frag_a_bf(const __bf16* __restrict__ A,
                                                       int lda, int row0, int k0) {
    int lane = threadIdx.x & 31;
    int half = lane >> 4;
    int m    = lane & 15;
    const __bf16* p = A + (size_t)(row0 + m) * lda + k0;
    v16bf a;
    __builtin_memcpy((char*)&a,      p + 8 * half,      16);
    __builtin_memcpy((char*)&a + 16, p + 16 + 8 * half, 16);
    return a;
}

static __device__ __forceinline__ v16bf load_frag_b_bf(const __bf16* __restrict__ W,
                                                       int ldw, int col0, int k0) {
    int lane = threadIdx.x & 31;
    int half = lane >> 4;
    int n    = lane & 15;
    const __bf16* p = W + (size_t)(col0 + n) * ldw + k0 + 8 * half;
    v16bf b;
    __builtin_memcpy((char*)&b,      p,      16);
    __builtin_memcpy((char*)&b + 16, p + 16, 16);
    return b;
}

enum { EPI_NONE = 0, EPI_BIAS = 1, EPI_TANH_BIAS = 2 };

// ---------------------------------------------------------------------------
// WMMA GEMM: C[m,n] = epi( sum_k A[m,k]*W[n,k] (+bias[n]) ); CT = f32 or bf16.
// One wave computes a 16x64 tile: 1 A-frag + 4 B-frags -> 4 WMMAs per k-step.
// ---------------------------------------------------------------------------
template <int EPI, typename CT>
__global__ __launch_bounds__(256)
void wmma_gemm(const __bf16* __restrict__ A, int lda,
               const __bf16* __restrict__ W, int ldw,
               const float* __restrict__ bias,
               CT* __restrict__ C, int ldc,
               int M, int N, int K) {
    int wid = (int)((blockIdx.x * blockDim.x + threadIdx.x) >> 5); // wave-uniform
    int ntn = N >> 6;
    int total = (M >> 4) * ntn;
    if (wid >= total) return; // whole-wave exit: EXEC all-1s for WMMA

    int tm = (wid / ntn) << 4;
    int tn = (wid % ntn) << 6;

    v8f acc0 = {}, acc1 = {}, acc2 = {}, acc3 = {};
    for (int k0 = 0; k0 < K; k0 += 32) {
        v16bf a  = load_frag_a_bf(A, lda, tm, k0);
        v16bf b0 = load_frag_b_bf(W, ldw, tn,      k0);
        v16bf b1 = load_frag_b_bf(W, ldw, tn + 16, k0);
        v16bf b2 = load_frag_b_bf(W, ldw, tn + 32, k0);
        v16bf b3 = load_frag_b_bf(W, ldw, tn + 48, k0);
        acc0 = __builtin_amdgcn_wmma_f32_16x16x32_bf16(false, a, false, b0, (short)0, acc0, false, false);
        acc1 = __builtin_amdgcn_wmma_f32_16x16x32_bf16(false, a, false, b1, (short)0, acc1, false, false);
        acc2 = __builtin_amdgcn_wmma_f32_16x16x32_bf16(false, a, false, b2, (short)0, acc2, false, false);
        acc3 = __builtin_amdgcn_wmma_f32_16x16x32_bf16(false, a, false, b3, (short)0, acc3, false, false);
    }

    int lane = threadIdx.x & 31;
    int half = lane >> 4;
    int n    = lane & 15;
    v8f accs[4] = {acc0, acc1, acc2, acc3};
#pragma unroll
    for (int j = 0; j < 4; ++j) {
        int col = tn + 16 * j + n;
        float bn = (EPI != EPI_NONE) ? bias[col] : 0.0f;
#pragma unroll
        for (int r = 0; r < 8; ++r) {
            int m = tm + r + 8 * half;
            float v = accs[j][r] + bn;
            if (EPI == EPI_TANH_BIAS) v = fast_tanh(v);
            C[(size_t)m * ldc + col] = (CT)v;
        }
    }
}

// ---------------------------------------------------------------------------
// Attention scores: one wave per (b,s). q[b] row staged via async->LDS and
// shared by all 8 waves of the block (a block never crosses a b boundary:
// S/8 = 64 blocks per b). pk is bf16, each lane owns 16 contiguous h.
// ---------------------------------------------------------------------------
__global__ __launch_bounds__(256)
void attn_scores_kernel(const float* __restrict__ q,       // B x H
                        const __bf16* __restrict__ pk,     // (B*S) x H bf16
                        const float* __restrict__ v_att,   // H
                        const unsigned char* __restrict__ mask, // B x S
                        float* __restrict__ scores) {      // B x S
    __shared__ float ql[H];
    int wave = threadIdx.x >> 5, lane = threadIdx.x & 31;
    int wid = blockIdx.x * 8 + wave;          // (b,s) row index
    int b = wid >> 9;                          // / S

    for (int i = threadIdx.x; i < H; i += 256)
        async_load_f32_to_lds(&ql[i], &q[(size_t)b * H + i]);
    wait_asynccnt0();
    __syncthreads();

    int h0 = lane * 16;
    const unsigned* pkr = (const unsigned*)(pk + (size_t)wid * H + h0);
    float sum = 0.0f;
#pragma unroll
    for (int j = 0; j < 8; ++j) {
        unsigned u = pkr[j];
        int h = h0 + 2 * j;
        sum += v_att[h]     * fast_tanh(ql[h]     + bf_lo(u));
        sum += v_att[h + 1] * fast_tanh(ql[h + 1] + bf_hi(u));
    }
#pragma unroll
    for (int off = 16; off > 0; off >>= 1)
        sum += __shfl_xor(sum, off, 32);
    if (lane == 0) {
        int s = wid & (S - 1);
        scores[wid] = mask[(size_t)b * S + s] ? sum : -1.0e9f;
    }
}

__global__ __launch_bounds__(512)
void softmax_kernel(float* __restrict__ scores) {
    __shared__ float redm[16];
    __shared__ float reds[16];
    int b = blockIdx.x;
    int tid = threadIdx.x;
    int wave = tid >> 5, lane = tid & 31;
    float x = scores[(size_t)b * S + tid];

    float m = x;
#pragma unroll
    for (int off = 16; off > 0; off >>= 1) m = fmaxf(m, __shfl_xor(m, off, 32));
    if (lane == 0) redm[wave] = m;
    __syncthreads();
    if (tid < 16) {
        float t = redm[tid];
#pragma unroll
        for (int off = 8; off > 0; off >>= 1) t = fmaxf(t, __shfl_xor(t, off, 16));
        redm[tid] = t;
    }
    __syncthreads();
    m = redm[0];

    float e = __expf(x - m);
    float ssum = e;
#pragma unroll
    for (int off = 16; off > 0; off >>= 1) ssum += __shfl_xor(ssum, off, 32);
    if (lane == 0) reds[wave] = ssum;
    __syncthreads();
    if (tid < 16) {
        float t = reds[tid];
#pragma unroll
        for (int off = 8; off > 0; off >>= 1) t += __shfl_xor(t, off, 16);
        reds[tid] = t;
    }
    __syncthreads();
    scores[(size_t)b * S + tid] = e * fast_rcp(reds[0]);
}

// context[b,d] = sum_s alphas[b,s]*enc_bf[b,s,d]; enc in bf16 (L2-resident),
// alphas staged via async->LDS; each thread owns a bf16 pair (2 d values).
__global__ __launch_bounds__(256)
void context_kernel(const float* __restrict__ alphas,    // B x S
                    const __bf16* __restrict__ enc_bf,   // B x S x 2H bf16
                    float* __restrict__ ctx) {           // B x 2H
    __shared__ float al[S];
    int chunks = K2H / 512;            // 2 (each block covers 512 d)
    int b = blockIdx.x / chunks;
    int chunk = blockIdx.x % chunks;
    int tid = threadIdx.x;
    for (int s = tid; s < S; s += 256)
        async_load_f32_to_lds(&al[s], &alphas[(size_t)b * S + s]);
    wait_asynccnt0();
    __syncthreads();

    int d0 = (chunk * 256 + tid) * 2;
    const unsigned* encb = (const unsigned*)(enc_bf + (size_t)b * S * K2H + d0);
    float acc0 = 0.0f, acc1 = 0.0f;
#pragma unroll 4
    for (int s = 0; s < S; ++s) {
        unsigned u = encb[(size_t)s * (K2H / 2)];
        float a = al[s];
        acc0 += a * bf_lo(u);
        acc1 += a * bf_hi(u);
    }
    ctx[(size_t)b * K2H + d0]     = acc0;
    ctx[(size_t)b * K2H + d0 + 1] = acc1;
}

// ---------------------------------------------------------------------------
// Packing / conversion / pointwise kernels
// ---------------------------------------------------------------------------
__global__ void cvt_f32_bf16_kernel(const float* __restrict__ src,
                                    __bf16* __restrict__ dst, int n) {
    int i = blockIdx.x * blockDim.x + threadIdx.x;
    if (i < n) dst[i] = (__bf16)src[i];
}

__global__ void pack_init_kernel(const float* __restrict__ enc_hn,
                                 const float* __restrict__ enc_cn,
                                 __bf16* __restrict__ A_hn,
                                 __bf16* __restrict__ A_cn) {
    int i = blockIdx.x * blockDim.x + threadIdx.x;
    if (i >= B * K2H) return;
    int b = i / K2H, k = i - b * K2H;
    float vh = (k < H) ? enc_hn[(size_t)b * H + k]
                       : enc_hn[(size_t)B * H + (size_t)b * H + (k - H)];
    const float* c1 = enc_cn + (size_t)B * H;
    float vc = (k < H) ? c1[(size_t)b * H + k] : c1[(size_t)b * H + (k - H)];
    A_hn[i] = (__bf16)vh;
    A_cn[i] = (__bf16)vc;
}

__global__ void build_wcat_kernel(const float* __restrict__ W_ih,
                                  const float* __restrict__ W_hh,
                                  __bf16* __restrict__ W_cat) {
    int i = blockIdx.x * blockDim.x + threadIdx.x;
    if (i >= N4H * KCAT) return;
    int n = i / KCAT, k = i - n * KCAT;
    float v = (k < E + 2 * H) ? W_ih[(size_t)n * (E + 2 * H) + k]
                              : W_hh[(size_t)n * H + (k - (E + 2 * H))];
    W_cat[i] = (__bf16)v;
}

__global__ void bias_sum_kernel(const float* __restrict__ b_ih,
                                const float* __restrict__ b_hh,
                                float* __restrict__ bsum) {
    int i = blockIdx.x * blockDim.x + threadIdx.x;
    if (i < N4H) bsum[i] = b_ih[i] + b_hh[i];
}

__global__ void pack_rnn_in_kernel(const float* __restrict__ embed_t,
                                   const float* __restrict__ ctx,
                                   const float* __restrict__ hn,
                                   __bf16* __restrict__ out) {
    int i = blockIdx.x * blockDim.x + threadIdx.x;
    if (i >= B * KCAT) return;
    int b = i / KCAT, k = i - b * KCAT;
    float v;
    if (k < E)               v = embed_t[(size_t)b * T * E + k];
    else if (k < E + 2 * H)  v = ctx[(size_t)b * K2H + (k - E)];
    else                     v = hn[(size_t)b * H + (k - E - 2 * H)];
    out[i] = (__bf16)v;
}

__global__ void pack_pre_in_kernel(const float* __restrict__ embed_t,
                                   const float* __restrict__ hn,
                                   const float* __restrict__ ctx,
                                   __bf16* __restrict__ out) {
    int i = blockIdx.x * blockDim.x + threadIdx.x;
    if (i >= B * KCAT) return;
    int b = i / KCAT, k = i - b * KCAT;
    float v;
    if (k < E)           v = embed_t[(size_t)b * T * E + k];
    else if (k < E + H)  v = hn[(size_t)b * H + (k - E)];
    else                 v = ctx[(size_t)b * K2H + (k - E - H)];
    out[i] = (__bf16)v;
}

__global__ void lstm_kernel(const float* __restrict__ gates,
                            float* __restrict__ hn,
                            __bf16* __restrict__ hn_bf,
                            float* __restrict__ cn,
                            float* __restrict__ dec_states_t) {
    int i = blockIdx.x * blockDim.x + threadIdx.x;
    if (i >= B * H) return;
    int b = i / H, h = i - b * H;
    const float* g = gates + (size_t)b * N4H;
    float ig = fast_sigmoid(g[h]);
    float fg = fast_sigmoid(g[H + h]);
    float gg = fast_tanh(g[2 * H + h]);
    float og = fast_sigmoid(g[3 * H + h]);
    float c  = fg * cn[i] + ig * gg;
    float hh = og * fast_tanh(c);
    cn[i] = c;
    hn[i] = hh;
    hn_bf[i] = (__bf16)hh;
    dec_states_t[(size_t)b * (T * H) + h] = hh;
}

__global__ void copy_states_kernel(const float* __restrict__ hn,
                                   const float* __restrict__ cn,
                                   float* __restrict__ out_hn,
                                   float* __restrict__ out_cn) {
    int i = blockIdx.x * blockDim.x + threadIdx.x;
    if (i >= B * H) return;
    out_hn[i] = hn[i];
    out_cn[i] = cn[i];
}

// ---------------------------------------------------------------------------
static inline int cdiv(int a, int b) { return (a + b - 1) / b; }

extern "C" void kernel_launch(void* const* d_in, const int* in_sizes, int n_in,
                              void* d_out, int out_size, void* d_ws, size_t ws_size,
                              hipStream_t stream) {
    const float* trg_embed  = (const float*)d_in[0];
    const float* enc_hidden = (const float*)d_in[1];
    const float* enc_hn     = (const float*)d_in[2];
    const float* enc_cn     = (const float*)d_in[3];
    const unsigned char* src_mask = (const unsigned char*)d_in[4];
    const float* Wk    = (const float*)d_in[7];
    const float* Wq    = (const float*)d_in[8];
    const float* v_att = (const float*)d_in[9];
    const float* W_ih  = (const float*)d_in[10];
    const float* W_hh  = (const float*)d_in[11];
    const float* b_ih  = (const float*)d_in[12];
    const float* b_hh  = (const float*)d_in[13];
    const float* W_bhn = (const float*)d_in[14];
    const float* b_bhn = (const float*)d_in[15];
    const float* W_bcn = (const float*)d_in[16];
    const float* b_bcn = (const float*)d_in[17];
    const float* W_pre = (const float*)d_in[18];

    // Output layout: dec_states | hnT | cnT | pre_outputs
    float* out_dec = (float*)d_out;
    float* out_hn  = out_dec + (size_t)B * T * H;
    float* out_cn  = out_hn + (size_t)B * H;
    float* out_pre = out_cn + (size_t)B * H;

    // Workspace: small f32 region, then bf16 region (~117 MB total).
    // pk_bf (32MB) + enc_bf (64MB) = 96MB hot per-step set -> L2-resident.
    float* wf = (float*)d_ws;
    float* q      = wf; wf += (size_t)B * H;
    float* scores = wf; wf += (size_t)B * S;
    float* ctx    = wf; wf += (size_t)B * K2H;
    float* gates  = wf; wf += (size_t)B * N4H;
    float* hn     = wf; wf += (size_t)B * H;
    float* cn     = wf; wf += (size_t)B * H;
    float* bsum   = wf; wf += (size_t)N4H;

    __bf16* wb = (__bf16*)wf;
    __bf16* pk_bf    = wb; wb += (size_t)B * S * H;     // 16.8M
    __bf16* enc_bf   = wb; wb += (size_t)B * S * K2H;   // 33.6M
    __bf16* Wcat_bf  = wb; wb += (size_t)N4H * KCAT;    // 3.67M
    __bf16* Wk_bf    = wb; wb += (size_t)H * K2H;
    __bf16* Wq_bf    = wb; wb += (size_t)H * H;
    __bf16* Wpre_bf  = wb; wb += (size_t)H * KCAT;
    __bf16* Wbhn_bf  = wb; wb += (size_t)H * K2H;
    __bf16* Wbcn_bf  = wb; wb += (size_t)H * K2H;
    __bf16* A_hn_bf  = wb; wb += (size_t)B * K2H;
    __bf16* A_cn_bf  = wb; wb += (size_t)B * K2H;
    __bf16* hn_bf    = wb; wb += (size_t)B * H;
    __bf16* rnn_in   = wb; wb += (size_t)B * KCAT;
    __bf16* pre_in   = wb; wb += (size_t)B * KCAT;

    const int BLK = 256;
    const int WPB = BLK / 32;

    // --- One-time prep ---
    cvt_f32_bf16_kernel<<<cdiv(B * S * K2H, BLK), BLK, 0, stream>>>(
        enc_hidden, enc_bf, B * S * K2H);
    build_wcat_kernel<<<cdiv(N4H * KCAT, BLK), BLK, 0, stream>>>(W_ih, W_hh, Wcat_bf);
    cvt_f32_bf16_kernel<<<cdiv(H * K2H, BLK), BLK, 0, stream>>>(Wk, Wk_bf, H * K2H);
    cvt_f32_bf16_kernel<<<cdiv(H * H, BLK), BLK, 0, stream>>>(Wq, Wq_bf, H * H);
    cvt_f32_bf16_kernel<<<cdiv(H * KCAT, BLK), BLK, 0, stream>>>(W_pre, Wpre_bf, H * KCAT);
    cvt_f32_bf16_kernel<<<cdiv(H * K2H, BLK), BLK, 0, stream>>>(W_bhn, Wbhn_bf, H * K2H);
    cvt_f32_bf16_kernel<<<cdiv(H * K2H, BLK), BLK, 0, stream>>>(W_bcn, Wbcn_bf, H * K2H);
    pack_init_kernel<<<cdiv(B * K2H, BLK), BLK, 0, stream>>>(enc_hn, enc_cn, A_hn_bf, A_cn_bf);
    bias_sum_kernel<<<cdiv(N4H, BLK), BLK, 0, stream>>>(b_ih, b_hh, bsum);

    // hn0 / cn0 (tanh epilogue), then bf16 copy of hn for the first q-GEMM
    {
        int tiles = (B / 16) * (H / 64);
        wmma_gemm<EPI_TANH_BIAS, float><<<cdiv(tiles, WPB), BLK, 0, stream>>>(
            A_hn_bf, K2H, Wbhn_bf, K2H, b_bhn, hn, H, B, H, K2H);
        wmma_gemm<EPI_TANH_BIAS, float><<<cdiv(tiles, WPB), BLK, 0, stream>>>(
            A_cn_bf, K2H, Wbcn_bf, K2H, b_bcn, cn, H, B, H, K2H);
    }
    cvt_f32_bf16_kernel<<<cdiv(B * H, BLK), BLK, 0, stream>>>(hn, hn_bf, B * H);

    // proj_key = encoder_hidden @ Wk^T, written directly as bf16
    {
        int tiles = ((B * S) / 16) * (H / 64); // 16384
        wmma_gemm<EPI_NONE, __bf16><<<cdiv(tiles, WPB), BLK, 0, stream>>>(
            enc_bf, K2H, Wk_bf, K2H, nullptr, pk_bf, H, B * S, H, K2H);
    }

    // --- Sequential decode scan ---
    for (int t = 0; t < T; ++t) {
        const float* embed_t = trg_embed + (size_t)t * E;

        // q = hn @ Wq^T
        {
            int tiles = (B / 16) * (H / 64);
            wmma_gemm<EPI_NONE, float><<<cdiv(tiles, WPB), BLK, 0, stream>>>(
                hn_bf, H, Wq_bf, H, nullptr, q, H, B, H, H);
        }
        attn_scores_kernel<<<(B * S) / 8, BLK, 0, stream>>>(
            q, pk_bf, v_att, src_mask, scores);
        softmax_kernel<<<B, S, 0, stream>>>(scores);
        context_kernel<<<B * (K2H / 512), 256, 0, stream>>>(scores, enc_bf, ctx);

        // gates = [embed|ctx|hn] @ W_cat^T + (b_ih + b_hh)
        pack_rnn_in_kernel<<<cdiv(B * KCAT, BLK), BLK, 0, stream>>>(embed_t, ctx, hn, rnn_in);
        {
            int tiles = (B / 16) * (N4H / 64);
            wmma_gemm<EPI_BIAS, float><<<cdiv(tiles, WPB), BLK, 0, stream>>>(
                rnn_in, KCAT, Wcat_bf, KCAT, bsum, gates, N4H, B, N4H, KCAT);
        }
        lstm_kernel<<<cdiv(B * H, BLK), BLK, 0, stream>>>(
            gates, hn, hn_bf, cn, out_dec + (size_t)t * H);

        // pre = [embed|hn_new|ctx] @ W_pre^T -> pre_outputs[:, t, :]
        pack_pre_in_kernel<<<cdiv(B * KCAT, BLK), BLK, 0, stream>>>(embed_t, hn, ctx, pre_in);
        {
            int tiles = (B / 16) * (H / 64);
            wmma_gemm<EPI_NONE, float><<<cdiv(tiles, WPB), BLK, 0, stream>>>(
                pre_in, KCAT, Wpre_bf, KCAT, nullptr, out_pre + (size_t)t * H, T * H, B, H, KCAT);
        }
    }

    copy_states_kernel<<<cdiv(B * H, BLK), BLK, 0, stream>>>(hn, cn, out_hn, out_cn);
}
